// RelativeMultiheadAttention_18494129177254
// MI455X (gfx1250) — matrix operand order repaired and checked by reading, not compiled, not run on hardware
//
#include <hip/hip_runtime.h>

typedef __attribute__((ext_vector_type(16))) __bf16 v16bf;
typedef __attribute__((ext_vector_type(8)))  float  v8f;

#define EMB   1024
#define NH    16
#define DK    64
#define SEQL  1024
#define MROWS 4096      // BATCH*SEQ

#define BM 128
#define BN 128
#define BK 32
#define LDP 40          // LDS row pitch in bf16 units (conflict-free for b128 reads)

__device__ __forceinline__ int lane_id() { return (int)(threadIdx.x & 31); }

// A-operand fragment (16x32 bf16): lane row = l%16; K pairs {0..7,16..23} (lanes<16)
// or {8..15,24..31} (lanes>=16), i.e. two b128 chunks at kofs+half*8 and +16.
__device__ __forceinline__ v16bf ld_frag_a(const __bf16* p, int stride, int row_base, int kofs) {
  int l = lane_id(); int hf = (l >> 4) & 1;
  const __bf16* b = p + (long)(row_base + (l & 15)) * stride + kofs + hf * 8;
  union { uint4 u[2]; v16bf v; } t;
  t.u[0] = *(const uint4*)(b);
  t.u[1] = *(const uint4*)(b + 16);
  return t.v;
}

// B-operand fragment (32x16 bf16): lane column = l%16; K {0..7,16..23} (lanes<16)
// or {8..15,24..31} (lanes>=16), i.e. two b128 chunks at kofs+half*8 and +16.
// dir=+1 normal rows, dir=-1 for the reversed relative-position band.
__device__ __forceinline__ v16bf ld_frag_b(const __bf16* p, int stride, int row_base, int dir, int kofs) {
  int l = lane_id(); int hf = (l >> 4) & 1;
  const __bf16* b = p + (long)(row_base + dir * (l & 15)) * stride + kofs + hf * 8;
  union { uint4 u[2]; v16bf v; } t;
  t.u[0] = *(const uint4*)(b);
  t.u[1] = *(const uint4*)(b + 16);
  return t.v;
}

__device__ __forceinline__ uint2 pack4(float a, float b, float c, float d) {
  union { __bf16 h[4]; uint2 u; } t;
  t.h[0] = (__bf16)a; t.h[1] = (__bf16)b; t.h[2] = (__bf16)c; t.h[3] = (__bf16)d;
  return t.u;
}

// ---------------------------------------------------------------------------
// rel_table f32 [2047,64] -> bf16 [2048,64] (row 2047 zero-padded)
// ---------------------------------------------------------------------------
__global__ void cvt_rel_table(const float* __restrict__ T, __bf16* __restrict__ Tb) {
  int i = blockIdx.x * 256 + threadIdx.x;        // 2048*64 elements
  if (i < 2048 * 64) {
    int r = i >> 6;
    float v = (r < 2047) ? T[i] : 0.0f;
    Tb[i] = (__bf16)v;
  }
}

// ---------------------------------------------------------------------------
// C[M,N] = A[M,K] @ B[N,K]^T + bias   (A,B f32 in HBM, bf16 in LDS, f32 accum)
// MODE 0: Cf f32 [M,N]
// MODE 1: Cb bf16 scattered to [b,h,n,d]   (Q,K)
// MODE 2: Cb bf16 scattered to [b,h,d,n]   (V transposed)
// M=4096, N=K=1024 fixed. Block tile 128x128x32, 8 waves of 32x64.
// ---------------------------------------------------------------------------
template <int MODE>
__global__ __launch_bounds__(256) void gemm_wmma(const float* __restrict__ A,
    const float* __restrict__ B, const float* __restrict__ bias,
    float* __restrict__ Cf, __bf16* __restrict__ Cb)
{
  __shared__ __bf16 sA[BM * LDP];
  __shared__ __bf16 sB[BN * LDP];

  const int tid  = threadIdx.x;
  const int l    = tid & 31;
  const int hf   = (l >> 4) & 1;
  const int wave = tid >> 5;
  const int wm   = wave & 3;          // 4 wave rows x 32
  const int wn   = wave >> 2;         // 2 wave cols x 64
  const int nb   = EMB / BN;          // 8
  const int bm0  = (blockIdx.x / nb) * BM;
  const int bn0  = (blockIdx.x % nb) * BN;

  const int ldr = tid >> 3;           // loader: row 0..127 per 32-element pass
  const int ldc = (tid & 7) * 4;      // loader: col (4 floats)

  v8f acc[2][4] = {};

  for (int k0 = 0; k0 < EMB; k0 += BK) {
    __syncthreads();
    // A tile 128x32 and B tile 128x32: f32 float4 loads -> packed bf16 -> b64 LDS
    for (int ps = 0; ps < 4; ++ps) {
      int r = ps * 32 + ldr;
      float4 va = *(const float4*)(A + (long)(bm0 + r) * EMB + k0 + ldc);
      *(uint2*)&sA[r * LDP + ldc] = pack4(va.x, va.y, va.z, va.w);
      float4 vb = *(const float4*)(B + (long)(bn0 + r) * EMB + k0 + ldc);
      *(uint2*)&sB[r * LDP + ldc] = pack4(vb.x, vb.y, vb.z, vb.w);
      if (k0 + BK < EMB) {            // warm L0/L2 for the next K tile
        __builtin_prefetch(A + (long)(bm0 + r) * EMB + k0 + BK + ldc, 0, 1);
        __builtin_prefetch(B + (long)(bn0 + r) * EMB + k0 + BK + ldc, 0, 1);
      }
    }
    __syncthreads();

    v16bf af[2], bfr[4];
    for (int mt = 0; mt < 2; ++mt) af[mt]  = ld_frag_a(sA, LDP, wm * 32 + mt * 16, 0);
    for (int nt = 0; nt < 4; ++nt) bfr[nt] = ld_frag_b(sB, LDP, wn * 64 + nt * 16, 1, 0);
    for (int mt = 0; mt < 2; ++mt)
      for (int nt = 0; nt < 4; ++nt)
        acc[mt][nt] = __builtin_amdgcn_wmma_f32_16x16x32_bf16(
            false, af[mt], false, bfr[nt], (short)0, acc[mt][nt], false, false);
  }

  for (int mt = 0; mt < 2; ++mt)
    for (int nt = 0; nt < 4; ++nt)
      for (int v = 0; v < 8; ++v) {
        int row = bm0 + wm * 32 + mt * 16 + v + hf * 8;   // D layout: vgpr=m%8, lane half = m/8
        int col = bn0 + wn * 64 + nt * 16 + (l & 15);
        float val = acc[mt][nt][v] + bias[col];
        if (MODE == 0) {
          Cf[(long)row * EMB + col] = val;
        } else if (MODE == 1) {
          long dst = (long)((row >> 10) * NH + (col >> 6)) * (SEQL * DK)
                   + (long)(row & 1023) * DK + (col & 63);
          Cb[dst] = (__bf16)val;
        } else {
          long dst = (long)((row >> 10) * NH + (col >> 6)) * (SEQL * DK)
                   + (long)(col & 63) * SEQL + (row & 1023);
          Cb[dst] = (__bf16)val;
        }
      }
}

// ---------------------------------------------------------------------------
// Fused flash-attention with relative position bias.
// One wave per (b,h, 16-row query tile). grid = 64 tiles * 64 bh = 4096.
// ---------------------------------------------------------------------------
__global__ __launch_bounds__(32) void attn_fused(const __bf16* __restrict__ Qb,
    const __bf16* __restrict__ Kb, const __bf16* __restrict__ Vtb,
    const __bf16* __restrict__ Tb, float* __restrict__ Oout)
{
  __shared__ __bf16 sP[16 * LDP];          // probs staging for D->A relayout

  const int l  = threadIdx.x & 31;
  const int n  = l & 15;
  const int hf = l >> 4;
  const int it = blockIdx.x & 63;
  const int bh = blockIdx.x >> 6;
  const int i0 = it * 16;

  const __bf16* Qh = Qb  + (long)bh * SEQL * DK;
  const __bf16* Kh = Kb  + (long)bh * SEQL * DK;
  const __bf16* Vh = Vtb + (long)bh * SEQL * DK;   // [d, n] layout

  v16bf qf[2];
  qf[0] = ld_frag_a(Qh, DK, i0, 0);
  qf[1] = ld_frag_a(Qh, DK, i0, 32);

  v8f O[4] = {};
  float mrow[8], lrow[8];
  for (int v = 0; v < 8; ++v) { mrow[v] = -3.0e38f; lrow[v] = 0.0f; }

  const float SCALE = 0.18033688011112042f;   // log2(e)/sqrt(64): work in exp2 domain

  for (int kt = 0; kt < SEQL / 32; ++kt) {
    const int j0 = kt * 32;
    v8f S[2];
    for (int s = 0; s < 2; ++s) {
      const int j0s = j0 + s * 16;
      // content scores: Q(16x64) . K^T  -> 2 WMMAs
      v8f c = {};
      for (int df = 0; df < 2; ++df) {
        v16bf kb = ld_frag_b(Kh, DK, j0s, 1, df * 32);
        c = __builtin_amdgcn_wmma_f32_16x16x32_bf16(false, qf[df], false, kb, (short)0, c, false, false);
      }
      // relative-position band: M1[ii,j']=Q.T[base-j'], M2[ii,j']=Q.T[base+16-j']
      const int base = i0 - j0s + 1023;        // band rows stay within [0, 2047]
      v8f m1 = {}, m2 = {};
      for (int df = 0; df < 2; ++df) {
        v16bf b1 = ld_frag_b(Tb, DK, base,      -1, df * 32);
        m1 = __builtin_amdgcn_wmma_f32_16x16x32_bf16(false, qf[df], false, b1, (short)0, m1, false, false);
        v16bf b2 = ld_frag_b(Tb, DK, base + 16, -1, df * 32);
        m2 = __builtin_amdgcn_wmma_f32_16x16x32_bf16(false, qf[df], false, b2, (short)0, m2, false, false);
      }
      // S_pos[m,n] = M1[m,(n-m)&15] if m<=n else M2[m,(n-m)&15] (same half, same vgpr)
      for (int v = 0; v < 8; ++v) {
        int m = v + hf * 8;
        int srcl = ((n - m) & 15) | (hf << 4);
        float p1 = __shfl(m1[v], srcl, 32);
        float p2 = __shfl(m2[v], srcl, 32);
        float sp = (m <= n) ? p1 : p2;
        S[s][v] = (c[v] + sp) * SCALE;
      }
    }

    // online softmax over these 32 keys (row = 16 lanes within a half)
    for (int v = 0; v < 8; ++v) {
      float t = fmaxf(S[0][v], S[1][v]);
      t = fmaxf(t, __shfl_xor(t, 1, 32));
      t = fmaxf(t, __shfl_xor(t, 2, 32));
      t = fmaxf(t, __shfl_xor(t, 4, 32));
      t = fmaxf(t, __shfl_xor(t, 8, 32));
      float mn = fmaxf(mrow[v], t);
      float al = exp2f(mrow[v] - mn);
      mrow[v] = mn;
      float p0 = exp2f(S[0][v] - mn);
      float p1 = exp2f(S[1][v] - mn);
      float rs = p0 + p1;
      rs += __shfl_xor(rs, 1, 32);
      rs += __shfl_xor(rs, 2, 32);
      rs += __shfl_xor(rs, 4, 32);
      rs += __shfl_xor(rs, 8, 32);
      lrow[v] = lrow[v] * al + rs;
      for (int t4 = 0; t4 < 4; ++t4) O[t4][v] *= al;
      int m = v + hf * 8;
      sP[m * LDP + n]      = (__bf16)p0;     // D layout -> LDS [m][key] bf16
      sP[m * LDP + 16 + n] = (__bf16)p1;
    }

    // P(16x32) . V(32x64): A from LDS, B straight from L2-resident V^T
    v16bf pA = ld_frag_a(sP, LDP, 0, 0);
    for (int dt = 0; dt < 4; ++dt) {
      v16bf bv = ld_frag_b(Vh, SEQL, dt * 16, 1, j0);
      O[dt] = __builtin_amdgcn_wmma_f32_16x16x32_bf16(false, pA, false, bv, (short)0, O[dt], false, false);
    }
  }

  // epilogue: normalize and scatter to [b, tok, h*64+d] (f32 for final GEMM)
  const int b = bh >> 4, h = bh & 15;
  for (int dt = 0; dt < 4; ++dt)
    for (int v = 0; v < 8; ++v) {
      int m = v + hf * 8;
      float val = O[dt][v] / lrow[v];
      long dst = (long)(b * SEQL + i0 + m) * EMB + h * DK + dt * 16 + n;
      Oout[dst] = val;
    }
}

// ---------------------------------------------------------------------------
extern "C" void kernel_launch(void* const* d_in, const int* in_sizes, int n_in,
                              void* d_out, int out_size, void* d_ws, size_t ws_size,
                              hipStream_t stream)
{
  (void)in_sizes; (void)n_in; (void)out_size; (void)ws_size;
  const float* x  = (const float*)d_in[0];
  const float* Wq = (const float*)d_in[1];
  const float* bq = (const float*)d_in[2];
  const float* Wk = (const float*)d_in[3];
  const float* bk = (const float*)d_in[4];
  const float* Wv = (const float*)d_in[5];
  const float* bv = (const float*)d_in[6];
  const float* Wo = (const float*)d_in[7];
  const float* bo = (const float*)d_in[8];
  const float* Tt = (const float*)d_in[9];
  float* out = (float*)d_out;

  char* w = (char*)d_ws;                                     // ws layout (40.5 MB):
  __bf16* Qb   = (__bf16*)(w);                               //  0..8MB   Q bf16 [b,h,n,d]
  __bf16* Kb   = (__bf16*)(w + (size_t)8  * 1024 * 1024);    //  8..16MB  K bf16 [b,h,n,d]
  __bf16* Vtb  = (__bf16*)(w + (size_t)16 * 1024 * 1024);    // 16..24MB  V bf16 [b,h,d,n]
  __bf16* Tb   = (__bf16*)(w + (size_t)24 * 1024 * 1024);    // 256KB rel_table bf16 (padded)
  float*  attnF = (float*)(w + (size_t)24 * 1024 * 1024 + 512 * 1024);  // 16MB attn out f32

  hipLaunchKernelGGL(cvt_rel_table, dim3(512), dim3(256), 0, stream, Tt, Tb);

  dim3 ggrid((MROWS / BM) * (EMB / BN));   // 32*8 = 256 workgroups
  hipLaunchKernelGGL((gemm_wmma<1>), ggrid, dim3(256), 0, stream, x, Wq, bq, (float*)nullptr, Qb);
  hipLaunchKernelGGL((gemm_wmma<1>), ggrid, dim3(256), 0, stream, x, Wk, bk, (float*)nullptr, Kb);
  hipLaunchKernelGGL((gemm_wmma<2>), ggrid, dim3(256), 0, stream, x, Wv, bv, (float*)nullptr, Vtb);

  hipLaunchKernelGGL(attn_fused, dim3(4096), dim3(32), 0, stream, Qb, Kb, Vtb, Tb, attnF);

  hipLaunchKernelGGL((gemm_wmma<0>), ggrid, dim3(256), 0, stream, attnF, Wo, bo, out, (__bf16*)nullptr);
}
